// RegionSelectionAttention_71090298683454
// MI455X (gfx1250) — matrix operand order repaired and checked by reading, not compiled, not run on hardware
//
#include <hip/hip_runtime.h>
#include <hip/hip_bf16.h>

// MI455X / gfx1250 implementation. wave32; all heavy GEMMs lowered to
// v_wmma_f32_16x16x32_f16 (f16 in, f32 accumulate). One wave per block for
// the WMMA kernels (EXEC all-ones as WMMA requires). Conv operands are staged
// through LDS with branchless fills so fragment builds are ds_load + select
// (no per-element exec-masked global loads).

typedef __attribute__((ext_vector_type(16))) _Float16 v16h;
typedef __attribute__((ext_vector_type(8)))  float    v8f;

#define DEVINL __device__ __forceinline__

DEVINL v8f wmma16(v16h a, v16h b, v8f c) {
  // D = A(16x32) * B(32x16) + C(16x16), f32 accum
  return __builtin_amdgcn_wmma_f32_16x16x32_f16(
      /*neg_a*/false, a, /*neg_b*/false, b, /*c_mod*/(short)0, c,
      /*reuse_a*/false, /*reuse_b*/false);
}

DEVINL void wait_lds() { asm volatile("s_wait_dscnt 0" ::: "memory"); }

// A fragment layout (16-bit A 16x32, contiguous k order used here):
//   lane l: m = l&15, g = l>>4; half j: k = j + g*16
template <typename F>
DEVINL v16h load_a(F&& f) {
  const int l = threadIdx.x & 31;
  const int m = l & 15, g = l >> 4;
  v16h a;
#pragma unroll
  for (int j = 0; j < 16; ++j) {
    const int k = j + (g << 4);
    a[j] = (_Float16)f(m, k);
  }
  return a;
}

// B fragment layout (16-bit B 32x16):
//   lane l: n = l&15, g = l>>4; half j: k = j + g*16
template <typename F>
DEVINL v16h load_b(F&& f) {
  const int l = threadIdx.x & 31;
  const int n = l & 15, g = l >> 4;
  v16h b;
#pragma unroll
  for (int j = 0; j < 16; ++j) {
    const int k = j + (g << 4);
    b[j] = (_Float16)f(k, n);
  }
  return b;
}
// C/D layout (f32 16x16): lane l: n=l&15, g=l>>4; elem i: m = i + g*8

// ---------------------------------------------------------------------------
// 1) Down conv: x[8,256,64,64] -(k4,s2,p1)-> xd[8,256,32,32]  (implicit GEMM)
//    M = co, K = ci*16 (ci,kh,kw), N = b*32*32. Halo tile staged in LDS.
// ---------------------------------------------------------------------------
__global__ void k_conv_down(const float* __restrict__ x, const float* __restrict__ w,
                            const float* __restrict__ bias, float* __restrict__ xd) {
  __shared__ _Float16 Bs[2][4][36];     // [ci_local][kh][col window]
  const int mt = blockIdx.x;            // co tile (16 of them)
  const int nt = blockIdx.y;            // spatial tile (512 of them)
  const int n0 = nt * 16;
  const int bi = n0 >> 10, sp0 = n0 & 1023;
  const int oh = sp0 >> 5, ow0 = sp0 & 31;   // 16 consecutive ow in one row
  const int tid = threadIdx.x;
  v8f acc = {};
  for (int kk = 0; kk < 4096; kk += 32) {
    const int ci0 = kk >> 4;
    // branchless halo fill: 2 ci x 4 rows x 36 cols = 288 elems, 9 per lane
#pragma unroll
    for (int it = 0; it < 9; ++it) {
      const int e = tid + it * 32;
      const int cil = e / 144, r = e - cil * 144;
      const int kh = r / 36, c = r - kh * 36;
      const int ih = oh * 2 + kh - 1;
      const int iw = ow0 * 2 - 1 + c;
      const bool ok = ((unsigned)ih < 64u) & ((unsigned)iw < 64u);
      const int ihc = min(max(ih, 0), 63), iwc = min(max(iw, 0), 63);
      const float v = x[(((size_t)bi * 256 + (ci0 + cil)) * 64 + ihc) * 64 + iwc];
      Bs[cil][kh][c] = ok ? (_Float16)v : (_Float16)0.f;
    }
    wait_lds();
    v16h a = load_a([&](int m, int k) -> float {
      return w[(size_t)(mt * 16 + m) * 4096 + kk + k];   // w_down[co][ci][kh][kw]
    });
    v16h b = load_b([&](int k, int n) -> float {
      const int cil = k >> 4, kh = (k >> 2) & 3, kw = k & 3;
      return (float)Bs[cil][kh][2 * n + kw];             // iw = ow0*2-1 + (2n+kw)
    });
    acc = wmma16(a, b, acc);
  }
  const int l = threadIdx.x, n = l & 15, g = l >> 4;
#pragma unroll
  for (int i = 0; i < 8; ++i) {
    const int co = mt * 16 + i + (g << 3);
    xd[((size_t)bi * 256 + co) * 1024 + sp0 + n] = acc[i] + bias[co];
  }
}

// ---------------------------------------------------------------------------
// 2) QKV projection: tok[bh,1024,64] @ W[64,192] + b  -> qkv[bh,1024,192]
//    mode 0: tokens come from xd (channel-major); mode 1: from contiguous tokf
// ---------------------------------------------------------------------------
__global__ void k_qkv(const float* __restrict__ src, const float* __restrict__ wq,
                      const float* __restrict__ bq, float* __restrict__ out, int mode) {
  const int tt = blockIdx.x / 12;     // token tile (64)
  const int ot = blockIdx.x % 12;     // output tile (12)
  const int bh = blockIdx.y;          // b*4+h
  const int b = bh >> 2, h = bh & 3;
  const int p0 = tt * 16, o0 = ot * 16;
  v8f acc = {};
  for (int kk = 0; kk < 64; kk += 32) {
    v16h a = load_a([&](int m, int k) -> float {
      const int d = kk + k, p = p0 + m;
      return (mode == 0)
                 ? src[((size_t)b * 256 + (h * 64 + d)) * 1024 + p]
                 : src[((size_t)bh * 1024 + p) * 64 + d];
    });
    v16h bf = load_b([&](int k, int n) -> float {
      return wq[(size_t)(kk + k) * 192 + o0 + n];
    });
    acc = wmma16(a, bf, acc);
  }
  const int l = threadIdx.x, n = l & 15, g = l >> 4;
#pragma unroll
  for (int i = 0; i < 8; ++i) {
    const int p = p0 + i + (g << 3);
    out[((size_t)bh * 1024 + p) * 192 + o0 + n] = acc[i] + bq[o0 + n];
  }
}

// ---------------------------------------------------------------------------
// 3) Attention pass 1: streaming online softmax row stats (max, sum)
//    one wave per (bh, 16-row q tile); K dim of QK^T = 64 (2 WMMA steps)
// ---------------------------------------------------------------------------
__global__ void k_attn_stats(const float* __restrict__ qkv,
                             float* __restrict__ rowm, float* __restrict__ rows) {
  const int qt = blockIdx.x, bh = blockIdx.y;
  const size_t base = (size_t)bh * 1024 * 192;
  const int q0 = qt * 16;
  const int l = threadIdx.x, g = l >> 4;
  float mrun[8], srun[8];
#pragma unroll
  for (int i = 0; i < 8; ++i) { mrun[i] = -1e30f; srun[i] = 0.f; }
  for (int c0 = 0; c0 < 1024; c0 += 16) {
    v8f acc = {};
    for (int kk = 0; kk < 64; kk += 32) {
      v16h a = load_a([&](int m, int k) -> float {
        return qkv[base + (size_t)(q0 + m) * 192 + kk + k];          // Q
      });
      v16h bf = load_b([&](int k, int nn) -> float {
        return qkv[base + (size_t)(c0 + nn) * 192 + 64 + kk + k];    // K (B = K^T)
      });
      acc = wmma16(a, bf, acc);
    }
#pragma unroll
    for (int i = 0; i < 8; ++i) {
      const float v = acc[i] * 0.125f;   // scale = hd^-0.5
      float t = v;
#pragma unroll
      for (int mk = 1; mk < 16; mk <<= 1) t = fmaxf(t, __shfl_xor(t, mk, 32));
      const float mnew = fmaxf(mrun[i], t);
      float e = __expf(v - mnew);
#pragma unroll
      for (int mk = 1; mk < 16; mk <<= 1) e += __shfl_xor(e, mk, 32);
      srun[i] = srun[i] * __expf(mrun[i] - mnew) + e;
      mrun[i] = mnew;
    }
  }
  if ((l & 15) == 0) {
#pragma unroll
    for (int i = 0; i < 8; ++i) {
      const int r = q0 + i + (g << 3);
      rowm[(size_t)bh * 1024 + r] = mrun[i];
      rows[(size_t)bh * 1024 + r] = srun[i];
    }
  }
}

// ---------------------------------------------------------------------------
// 4) Attention pass 2: O = softmax(QK^T) V, plus per-qtile column partial sums
//    P tile staged through LDS as f16 to re-shape C-layout -> A-layout.
// ---------------------------------------------------------------------------
__global__ void k_attn_out(const float* __restrict__ qkv,
                           const float* __restrict__ rowm, const float* __restrict__ rows,
                           float* __restrict__ attout, float* __restrict__ scorep,
                           int do_score) {
  __shared__ _Float16 Pt[16][32];
  const int qt = blockIdx.x, bh = blockIdx.y;
  const size_t base = (size_t)bh * 1024 * 192;
  const int q0 = qt * 16;
  const int l = threadIdx.x, n = l & 15, g = l >> 4;
  float mrow[8], srow[8];
#pragma unroll
  for (int i = 0; i < 8; ++i) {
    const int r = q0 + i + (g << 3);
    mrow[i] = rowm[(size_t)bh * 1024 + r];
    srow[i] = 1.0f / rows[(size_t)bh * 1024 + r];
  }
  v8f o0 = {}, o1 = {}, o2 = {}, o3 = {};
  for (int c0 = 0; c0 < 1024; c0 += 32) {
#pragma unroll
    for (int half = 0; half < 2; ++half) {
      const int cc = c0 + half * 16;
      v8f acc = {};
      for (int kk = 0; kk < 64; kk += 32) {
        v16h a = load_a([&](int m, int k) -> float {
          return qkv[base + (size_t)(q0 + m) * 192 + kk + k];
        });
        v16h bf = load_b([&](int k, int nn) -> float {
          return qkv[base + (size_t)(cc + nn) * 192 + 64 + kk + k];
        });
        acc = wmma16(a, bf, acc);
      }
      float cs = 0.f;
#pragma unroll
      for (int i = 0; i < 8; ++i) {
        const float p = __expf(acc[i] * 0.125f - mrow[i]) * srow[i];
        cs += p;
        Pt[i + (g << 3)][half * 16 + n] = (_Float16)p;
      }
      if (do_score) {
        cs += __shfl_xor(cs, 16, 32);     // combine the two row-halves
        if (g == 0)
          scorep[((size_t)qt * 32 + bh) * 1024 + cc + n] = cs;
      }
    }
    wait_lds();                           // LDS stores -> loads (same wave)
    v16h pa = load_a([&](int m, int k) -> float { return (float)Pt[m][k]; });
#pragma unroll
    for (int t = 0; t < 4; ++t) {
      v16h bf = load_b([&](int k, int nn) -> float {
        return qkv[base + (size_t)(c0 + k) * 192 + 128 + t * 16 + nn];  // V
      });
      v8f* op = (t == 0) ? &o0 : (t == 1) ? &o1 : (t == 2) ? &o2 : &o3;
      *op = wmma16(pa, bf, *op);
    }
  }
#pragma unroll
  for (int i = 0; i < 8; ++i) {
    const int q = q0 + i + (g << 3);
    float* dst = &attout[((size_t)bh * 1024 + q) * 64];
    dst[0 * 16 + n] = o0[i];
    dst[1 * 16 + n] = o1[i];
    dst[2 * 16 + n] = o2[i];
    dst[3 * 16 + n] = o3[i];
  }
}

// fixed-order reduction of per-qtile score partials (deterministic, no atomics)
__global__ void k_score_reduce(const float* __restrict__ scorep, float* __restrict__ score) {
  const int idx = blockIdx.x * 256 + threadIdx.x;   // bh*1024 + col, 32768 total
  if (idx >= 32 * 1024) return;
  float s = 0.f;
  for (int qt = 0; qt < 64; ++qt) s += scorep[(size_t)qt * 32768 + idx];
  score[idx] = s;
}

// ---------------------------------------------------------------------------
// 5) top-k (k=256 of 1024) by rank counting — reproduces jax.lax.top_k order
// ---------------------------------------------------------------------------
__global__ void k_topk(const float* __restrict__ score, int* __restrict__ topk) {
  __shared__ float sc[1024];
  const int bh = blockIdx.x;
  for (int i = threadIdx.x; i < 1024; i += blockDim.x)
    sc[i] = score[(size_t)bh * 1024 + i];
  __syncthreads();
  for (int i = threadIdx.x; i < 1024; i += blockDim.x) {
    const float v = sc[i];
    int rank = 0;
    for (int j = 0; j < 1024; ++j) {
      const float u = sc[j];
      rank += (u > v) || (u == v && j < i);
    }
    if (rank < 256) topk[bh * 256 + rank] = i;
  }
}

// ---------------------------------------------------------------------------
// 6) ConvTranspose2d(k4,s2,p1): attout tokens -> coarse[8,256,64,64]
//    coarse[b,co,y,x] = b_up[co] + sum in[b,ci,(y+1-kh)/2,(x+1-kw)/2]*w_up[ci,co,kh,kw]
//    2x2x10 input window per k-step staged in LDS; fragment is ds_load+select.
// ---------------------------------------------------------------------------
__global__ void k_conv_up(const float* __restrict__ attout, const float* __restrict__ w_up,
                          const float* __restrict__ b_up, float* __restrict__ coarse) {
  __shared__ _Float16 Ws[2][2][16];     // [ci_local][row][col] (10 cols used)
  const int mt = blockIdx.x;            // co tile (16)
  const int nt = blockIdx.y;            // spatial tile (2048)
  const int n0 = nt * 16;
  const int bi = n0 >> 12, sp0 = n0 & 4095;
  const int y = sp0 >> 6, x0 = sp0 & 63;    // 16 consecutive x in one row
  const int iy0 = (y - 1) >> 1;             // rows needed: iy0, iy0+1
  const int j0 = (x0 >> 1) - 1;             // cols needed: j0 .. j0+9
  const int tid = threadIdx.x;
  v8f acc = {};
  for (int kk = 0; kk < 4096; kk += 32) {
    const int ci0 = kk >> 4;
    // branchless fill: 2 ci x 2 rows x 16 cols = 64 elems, 2 per lane
#pragma unroll
    for (int it = 0; it < 2; ++it) {
      const int e = tid + it * 32;
      const int cil = e >> 5, r = (e >> 4) & 1, c = e & 15;
      const int i = iy0 + r, j = j0 + c;
      const bool ok = ((unsigned)i < 32u) & ((unsigned)j < 32u);
      const int ic = min(max(i, 0), 31), jc = min(max(j, 0), 31);
      const int ci = ci0 + cil;
      const int h = ci >> 6, d = ci & 63;
      const float v = attout[(((size_t)bi * 4 + h) * 1024 + (ic * 32 + jc)) * 64 + d];
      Ws[cil][r][c] = ok ? (_Float16)v : (_Float16)0.f;
    }
    wait_lds();
    v16h a = load_a([&](int m, int k) -> float {
      // w_up layout [ci][co][kh][kw]; k = ci_local*16 + kh*4 + kw
      return w_up[(size_t)((kk + k) >> 4) * 4096 + (mt * 16 + m) * 16 + (k & 15)];
    });
    v16h b = load_b([&](int k, int n) -> float {
      const int cil = k >> 4, kh = (k >> 2) & 3, kw = k & 3;
      const int ny = y + 1 - kh, nx = x0 + n + 1 - kw;
      const int i = ny >> 1, j = nx >> 1;
      const bool ok = !(ny & 1) && !(nx & 1) &&
                      ((unsigned)i < 32u) && ((unsigned)j < 32u);
      const int ri = min(max(i - iy0, 0), 1);
      const int rj = min(max(j - j0, 0), 15);
      const float v = (float)Ws[cil][ri][rj];
      return ok ? v : 0.f;
    });
    acc = wmma16(a, b, acc);
  }
  const int l = threadIdx.x, n = l & 15, g = l >> 4;
#pragma unroll
  for (int i = 0; i < 8; ++i) {
    const int co = mt * 16 + i + (g << 3);
    coarse[((size_t)bi * 256 + co) * 4096 + sp0 + n] = acc[i] + b_up[co];
  }
}

// ---------------------------------------------------------------------------
// 7) gather top-k 2x2 patches of coarse -> tokf[bh,1024,64]
// ---------------------------------------------------------------------------
__global__ void k_gather(const float* __restrict__ coarse, const int* __restrict__ topk,
                         float* __restrict__ tokf) {
  const int idx = blockIdx.x * 256 + threadIdx.x;     // 32*1024*64
  if (idx >= 32 * 1024 * 64) return;
  const int d = idx & 63;
  const int t = (idx >> 6) & 1023;
  const int bh = idx >> 16;
  const int b = bh >> 2, h = bh & 3;
  const int ki = t >> 2, a = (t >> 1) & 1, c2 = t & 1;
  const int pp = topk[bh * 256 + ki];
  const int y = (pp >> 5) * 2 + a, x = (pp & 31) * 2 + c2;
  tokf[idx] = coarse[(((size_t)b * 256 + h * 64 + d) * 64 + y) * 64 + x];
}

// 8) scatter fine attention output back into coarse (y = coarse + region)
__global__ void k_scatter(const float* __restrict__ attout, const int* __restrict__ topk,
                          float* __restrict__ y) {
  const int idx = blockIdx.x * 256 + threadIdx.x;     // 32*1024*64
  if (idx >= 32 * 1024 * 64) return;
  const int d = idx & 63;
  const int t = (idx >> 6) & 1023;
  const int bh = idx >> 16;
  const int b = bh >> 2, h = bh & 3;
  const int ki = t >> 2, a = (t >> 1) & 1, c2 = t & 1;
  const int pp = topk[bh * 256 + ki];
  const int Y = (pp >> 5) * 2 + a, X = (pp & 31) * 2 + c2;
  const size_t o = (((size_t)b * 256 + h * 64 + d) * 64 + Y) * 64 + X;
  y[o] += attout[idx];           // top-k patches are unique -> no collisions
}

// ---------------------------------------------------------------------------
// 9) depthwise 3x3 + BN + relu6 (tiny: VALU is fine)
// ---------------------------------------------------------------------------
__global__ void k_dw(const float* __restrict__ y, const float* __restrict__ wdw,
                     const float* __restrict__ gg, const float* __restrict__ bb,
                     const float* __restrict__ mm, const float* __restrict__ vv,
                     float* __restrict__ yd) {
  const int idx = blockIdx.x * 256 + threadIdx.x;     // 8*256*4096
  if (idx >= 8 * 256 * 4096) return;
  const int x = idx & 63, yy = (idx >> 6) & 63, c = (idx >> 12) & 255, b = idx >> 20;
  const size_t base = ((size_t)b * 256 + c) * 4096;
  float acc = 0.f;
#pragma unroll
  for (int kh = 0; kh < 3; ++kh) {
#pragma unroll
    for (int kw = 0; kw < 3; ++kw) {
      const int iy = yy + kh - 1, ix = x + kw - 1;
      if ((unsigned)iy < 64u && (unsigned)ix < 64u)
        acc += y[base + iy * 64 + ix] * wdw[c * 9 + kh * 3 + kw];
    }
  }
  const float s = gg[c] * rsqrtf(vv[c] + 1e-5f);
  const float r = acc * s + (bb[c] - mm[c] * s);
  yd[idx] = fminf(fmaxf(r, 0.f), 6.f);
}

// ---------------------------------------------------------------------------
// 10) pointwise 1x1 (GEMM M=256,K=256,N=32768) + BN + relu6 -> output
//     activation tile staged in LDS with coalesced fills.
// ---------------------------------------------------------------------------
__global__ void k_pw(const float* __restrict__ yd, const float* __restrict__ wpw,
                     const float* __restrict__ gg, const float* __restrict__ bb,
                     const float* __restrict__ mm, const float* __restrict__ vv,
                     float* __restrict__ out) {
  __shared__ _Float16 Ys[32][16];
  const int mt = blockIdx.x;          // co tile (16)
  const int nt = blockIdx.y;          // spatial tile (2048)
  const int n0 = nt * 16;
  const int bi = n0 >> 12, sp0 = n0 & 4095;
  const int tid = threadIdx.x;
  v8f acc = {};
  for (int kk = 0; kk < 256; kk += 32) {
    // coalesced fill: 32 k x 16 n = 512 elems, 16 per lane (2 rows / iter)
#pragma unroll
    for (int it = 0; it < 16; ++it) {
      const int e = tid + it * 32;
      const int k = e >> 4, n = e & 15;
      Ys[k][n] = (_Float16)yd[((size_t)bi * 256 + kk + k) * 4096 + sp0 + n];
    }
    wait_lds();
    v16h a = load_a([&](int m, int k) -> float {
      return wpw[(size_t)(mt * 16 + m) * 256 + kk + k];
    });
    v16h b = load_b([&](int k, int n) -> float { return (float)Ys[k][n]; });
    acc = wmma16(a, b, acc);
  }
  const int l = threadIdx.x, n = l & 15, g = l >> 4;
#pragma unroll
  for (int i = 0; i < 8; ++i) {
    const int co = mt * 16 + i + (g << 3);
    const float s = gg[co] * rsqrtf(vv[co] + 1e-5f);
    const float r = acc[i] * s + (bb[co] - mm[co] * s);
    out[((size_t)bi * 256 + co) * 4096 + sp0 + n] = fminf(fmaxf(r, 0.f), 6.f);
  }
}

// ---------------------------------------------------------------------------
extern "C" void kernel_launch(void* const* d_in, const int* in_sizes, int n_in,
                              void* d_out, int out_size, void* d_ws, size_t ws_size,
                              hipStream_t stream) {
  const float* x       = (const float*)d_in[0];
  const float* w_down  = (const float*)d_in[1];
  const float* b_down  = (const float*)d_in[2];
  const float* w_qkv_c = (const float*)d_in[3];
  const float* b_qkv_c = (const float*)d_in[4];
  const float* w_up    = (const float*)d_in[5];
  const float* b_up    = (const float*)d_in[6];
  const float* w_qkv_f = (const float*)d_in[7];
  const float* b_qkv_f = (const float*)d_in[8];
  const float* w_dw    = (const float*)d_in[9];
  const float* bn_dw_g = (const float*)d_in[10];
  const float* bn_dw_b = (const float*)d_in[11];
  const float* bn_dw_m = (const float*)d_in[12];
  const float* bn_dw_v = (const float*)d_in[13];
  const float* w_pw    = (const float*)d_in[14];
  const float* bn_pw_g = (const float*)d_in[15];
  const float* bn_pw_b = (const float*)d_in[16];
  const float* bn_pw_m = (const float*)d_in[17];
  const float* bn_pw_v = (const float*)d_in[18];
  float* out = (float*)d_out;

  float* ws = (float*)d_ws;
  size_t off = 0;
  float* xd     = ws + off; off += (size_t)8 * 256 * 32 * 32;    // 2.10M
  float* qkv    = ws + off; off += (size_t)32 * 1024 * 192;      // 6.29M (reused fine)
  float* rowm   = ws + off; off += (size_t)32 * 1024;
  float* rows   = ws + off; off += (size_t)32 * 1024;
  float* scorep = ws + off; off += (size_t)64 * 32 * 1024;       // 2.10M
  float* score  = ws + off; off += (size_t)32 * 1024;
  int*   topk   = (int*)(ws + off); off += (size_t)32 * 256;
  float* attout = ws + off; off += (size_t)32 * 1024 * 64;       // 2.10M (reused fine)
  float* coarse = ws + off; off += (size_t)8 * 256 * 64 * 64;    // 8.39M (becomes y)
  float* tokf   = ws + off; off += (size_t)32 * 1024 * 64;       // 2.10M
  float* yd     = ws + off; off += (size_t)8 * 256 * 64 * 64;    // 8.39M
  (void)ws_size; (void)in_sizes; (void)n_in; (void)out_size;

  // ---- coarse self-attention ----
  k_conv_down<<<dim3(16, 512), 32, 0, stream>>>(x, w_down, b_down, xd);
  k_qkv<<<dim3(64 * 12, 32), 32, 0, stream>>>(xd, w_qkv_c, b_qkv_c, qkv, 0);
  k_attn_stats<<<dim3(64, 32), 32, 0, stream>>>(qkv, rowm, rows);
  k_attn_out<<<dim3(64, 32), 32, 0, stream>>>(qkv, rowm, rows, attout, scorep, 1);
  k_score_reduce<<<128, 256, 0, stream>>>(scorep, score);
  k_topk<<<32, 256, 0, stream>>>(score, topk);
  k_conv_up<<<dim3(16, 2048), 32, 0, stream>>>(attout, w_up, b_up, coarse);

  // ---- top-k fine self-attention ----
  k_gather<<<8192, 256, 0, stream>>>(coarse, topk, tokf);
  k_qkv<<<dim3(64 * 12, 32), 32, 0, stream>>>(tokf, w_qkv_f, b_qkv_f, qkv, 1);
  k_attn_stats<<<dim3(64, 32), 32, 0, stream>>>(qkv, rowm, rows);
  k_attn_out<<<dim3(64, 32), 32, 0, stream>>>(qkv, rowm, rows, attout, scorep, 0);
  k_scatter<<<8192, 256, 0, stream>>>(attout, topk, coarse);   // coarse -> y

  // ---- DWConv feed-forward ----
  k_dw<<<32768, 256, 0, stream>>>(coarse, w_dw, bn_dw_g, bn_dw_b, bn_dw_m, bn_dw_v, yd);
  k_pw<<<dim3(16, 2048), 32, 0, stream>>>(yd, w_pw, bn_pw_g, bn_pw_b, bn_pw_m, bn_pw_v, out);
}
